// Encoder_38500086841910
// MI455X (gfx1250) — compile-verified, matrix operand order checked by reading, not run on hardware
//
#include <hip/hip_runtime.h>
#include <hip/hip_bf16.h>

#define N_   8
#define S_   1023
#define T_   1024
#define E_   768
#define H_   12
#define FF_  3072
#define L_   6
#define MR_  (N_ * T_)   // 8192 rows of activations

typedef __attribute__((ext_vector_type(16))) __bf16 v16bf;
typedef __attribute__((ext_vector_type(8)))  __bf16 v8bf;
typedef __attribute__((ext_vector_type(8)))  float  v8f;
typedef __attribute__((ext_vector_type(4)))  int    v4i;

// ---------------------------------------------------------------------------
// CDNA5 async global->LDS copy (ASYNCcnt-tracked), with sync fallback.
// Builtin expects: (v4i addrspace(1)*, v4i addrspace(3)*, imm offset, imm cpol)
// ---------------------------------------------------------------------------
#if defined(__HIP_DEVICE_COMPILE__) && \
    __has_builtin(__builtin_amdgcn_global_load_async_to_lds_b128) && \
    __has_builtin(__builtin_amdgcn_s_wait_asynccnt)
#define USE_ASYNC_COPY 1
#else
#define USE_ASYNC_COPY 0
#endif

#if USE_ASYNC_COPY
typedef __attribute__((address_space(1))) v4i as1_v4i;
typedef __attribute__((address_space(3))) v4i as3_v4i;
#endif

static __device__ __forceinline__ void async_ld_b128(__bf16* lds, const __bf16* g) {
#if USE_ASYNC_COPY
    __builtin_amdgcn_global_load_async_to_lds_b128(
        (as1_v4i*)g, (as3_v4i*)lds, 0, 0);
#else
    *(v8bf*)lds = *(const v8bf*)g;
#endif
}

template <int N>
static __device__ __forceinline__ void wait_async() {
#if USE_ASYNC_COPY
    __builtin_amdgcn_s_wait_asynccnt(N);
#endif
}

// Build a 16-element bf16 WMMA fragment from two contiguous 8-element runs.
// CDNA5 16-bit A/B layout: VGPRs 0-3 hold K=klo..klo+7, VGPRs 4-7 hold K=16+klo..+7.
static __device__ __forceinline__ v16bf frag16(const __bf16* p0, const __bf16* p1) {
    v8bf a = *(const v8bf*)p0;
    v8bf b = *(const v8bf*)p1;
    return __builtin_shufflevector(a, b, 0,1,2,3,4,5,6,7,8,9,10,11,12,13,14,15);
}

// ---------------------------------------------------------------------------
// Embed: out[n,t,:] = (t==0 ? cls : x[n,t-1,:]) + pos[t,:]   (fp32 + bf16 copy)
// ---------------------------------------------------------------------------
__global__ __launch_bounds__(256)
void embed_kernel(const float* __restrict__ x, const float* __restrict__ cls,
                  const float* __restrict__ pos,
                  float* __restrict__ act, __bf16* __restrict__ act16)
{
    size_t idx = (size_t)blockIdx.x * 256 + threadIdx.x;
    int e = idx % E_;
    int t = (idx / E_) % T_;
    int n = idx / ((size_t)E_ * T_);
    float v = (t == 0) ? cls[e] : x[((size_t)n * S_ + (t - 1)) * E_ + e];
    v += pos[(size_t)t * E_ + e];
    act[idx]   = v;
    act16[idx] = (__bf16)v;
}

// ---------------------------------------------------------------------------
// Transpose fp32 RxC -> bf16 CxR  (weight prep: B^T layout for fragment loads)
// ---------------------------------------------------------------------------
__global__ __launch_bounds__(256)
void transpose_w_kernel(const float* __restrict__ in, __bf16* __restrict__ out,
                        int R, int C)
{
    size_t idx = (size_t)blockIdx.x * 256 + threadIdx.x;
    int r = idx % R;          // original row (K dim)
    int c = idx / R;          // original col (N dim)
    out[idx] = (__bf16)in[(size_t)r * C + c];
}

// ---------------------------------------------------------------------------
// WMMA GEMM: C[M,Nn] = A[M,K](bf16) @ B (given as BT[Nn,K] bf16) + bias
// Block = 256 thr = 8 waves, block tile 128x128, wave tile 32x64, K-step 64,
// double-buffered LDS with async global->LDS prefetch (ASYNCcnt pipelined).
// gelu: 1 => exact GELU.  vtrans: 1 => write bf16 out in [N,H,64,T] layout.
// ---------------------------------------------------------------------------
#define LDA_ 72   // padded LDS row stride in halves (144B; spreads banks)

__global__ __launch_bounds__(256)
void gemm_bf16_kernel(const __bf16* __restrict__ A, const __bf16* __restrict__ BT,
                      const float* __restrict__ bias,
                      float* __restrict__ Cf, __bf16* __restrict__ Cb,
                      int M, int Nn, int K, int gelu, int vtrans)
{
    __shared__ __attribute__((aligned(16))) __bf16 As[2][128 * LDA_];
    __shared__ __attribute__((aligned(16))) __bf16 Bs[2][128 * LDA_];

    const int tid  = threadIdx.x;
    const int lane = tid & 31;
    const int wid  = tid >> 5;
    const int mw   = wid & 3;           // 0..3  -> 32-row group
    const int nw   = wid >> 2;          // 0..1  -> 64-col group
    const int rowBase = blockIdx.x * 128;
    const int colBase = blockIdx.y * 128;
    const int klo  = (lane & 16) ? 8 : 0;
    const int lrow = lane & 15;
    const int grp  = (lane >> 4) & 1;

    // staging coordinates: 128 rows x 64 halves per tile, 8 b128 chunks/row
    const int sr = tid >> 3;            // 0..31  (row step 32 over 4 iters)
    const int sc = (tid & 7) * 8;       // 0..56

    v8f c[2][4] = {};

    // ---- stage first K-tile into buffer 0 ----
    #pragma unroll
    for (int i = 0; i < 4; ++i) {
        int r = sr + i * 32;
        async_ld_b128(&As[0][r * LDA_ + sc], A  + (size_t)(rowBase + r) * K + sc);
        async_ld_b128(&Bs[0][r * LDA_ + sc], BT + (size_t)(colBase + r) * K + sc);
    }

    int p = 0;
    for (int k0 = 0; k0 < K; k0 += 64) {
        if (k0 + 64 < K) {
            // prefetch next K-tile into other buffer (overlaps with compute)
            #pragma unroll
            for (int i = 0; i < 4; ++i) {
                int r = sr + i * 32;
                async_ld_b128(&As[p ^ 1][r * LDA_ + sc],
                              A  + (size_t)(rowBase + r) * K + k0 + 64 + sc);
                async_ld_b128(&Bs[p ^ 1][r * LDA_ + sc],
                              BT + (size_t)(colBase + r) * K + k0 + 64 + sc);
            }
            wait_async<8>();   // current tile's 8 loads complete; next 8 in flight
        } else {
            wait_async<0>();
        }
        __syncthreads();

        #pragma unroll
        for (int ks = 0; ks < 2; ++ks) {        // two K=32 sub-steps
            const int co = ks * 32;
            v16bf a[2], b[4];
            #pragma unroll
            for (int mi = 0; mi < 2; ++mi) {
                int row = mw * 32 + mi * 16 + lrow;
                a[mi] = frag16(&As[p][row * LDA_ + co + klo],
                               &As[p][row * LDA_ + co + 16 + klo]);
            }
            #pragma unroll
            for (int ni = 0; ni < 4; ++ni) {
                int col = nw * 64 + ni * 16 + lrow;
                b[ni] = frag16(&Bs[p][col * LDA_ + co + klo],
                               &Bs[p][col * LDA_ + co + 16 + klo]);
            }
            #pragma unroll
            for (int mi = 0; mi < 2; ++mi)
                #pragma unroll
                for (int ni = 0; ni < 4; ++ni)
                    c[mi][ni] = __builtin_amdgcn_wmma_f32_16x16x32_bf16(
                        false, a[mi], false, b[ni], (short)0, c[mi][ni], false, false);
        }
        __syncthreads();   // all waves done reading buffer p before it is restaged
        p ^= 1;
    }

    // ---- epilogue: bias (+ GELU), scatter per C-matrix layout ----
    #pragma unroll
    for (int mi = 0; mi < 2; ++mi) {
        #pragma unroll
        for (int ni = 0; ni < 4; ++ni) {
            int col = colBase + nw * 64 + ni * 16 + lrow;
            float bv = bias ? bias[col] : 0.0f;
            #pragma unroll
            for (int r = 0; r < 8; ++r) {
                int row = rowBase + mw * 32 + mi * 16 + r + 8 * grp;
                float v = c[mi][ni][r] + bv;
                if (gelu) v = 0.5f * v * (1.0f + erff(v * 0.70710678118654752f));
                if (Cf) Cf[(size_t)row * Nn + col] = v;
                if (Cb) {
                    if (vtrans) {
                        // row = n*T + t ; col = h*64 + dd  ->  VT[N,H,64,T]
                        int n  = row >> 10, t = row & (T_ - 1);
                        int h  = col >> 6,  dd = col & 63;
                        Cb[(((size_t)n * H_ + h) * 64 + dd) * T_ + t] = (__bf16)v;
                    } else {
                        Cb[(size_t)row * Nn + col] = (__bf16)v;
                    }
                }
            }
        }
    }
}

// ---------------------------------------------------------------------------
// Flash attention: block = 128 thr (4 waves), each wave owns 16 queries.
// grid = (T/64, N*H). QK^T and PV both on v_wmma_f32_16x16x32_bf16.
// ---------------------------------------------------------------------------
__global__ __launch_bounds__(128)
void attn_kernel(const __bf16* __restrict__ Q, const __bf16* __restrict__ Kmat,
                 const __bf16* __restrict__ VT, const int* __restrict__ mask,
                 __bf16* __restrict__ O)
{
    __shared__ __attribute__((aligned(16))) __bf16 Plds[4][16][32];

    const int tid  = threadIdx.x;
    const int lane = tid & 31;
    const int w    = tid >> 5;
    const int nh   = blockIdx.y;
    const int n    = nh / H_;
    const int h    = nh % H_;
    const int qbase = blockIdx.x * 64 + w * 16;
    const int klo  = (lane & 16) ? 8 : 0;
    const int lrow = lane & 15;
    const int grp  = (lane >> 4) & 1;

    // Q fragments for the two d-halves (d=64 -> two K=32 steps)
    const __bf16* qrow = Q + ((size_t)n * T_ + qbase + lrow) * E_ + h * 64;
    v16bf aq[2];
    #pragma unroll
    for (int kk = 0; kk < 2; ++kk)
        aq[kk] = frag16(qrow + kk * 32 + klo, qrow + kk * 32 + 16 + klo);

    v8f o[4] = {};
    float mrow[8], lsum[8];
    #pragma unroll
    for (int r = 0; r < 8; ++r) { mrow[r] = -3.0e38f; lsum[r] = 0.0f; }

    const __bf16* Kbase = Kmat + (size_t)n * T_ * E_ + h * 64;
    const __bf16* Vbase = VT + (size_t)nh * 64 * T_;
    const int*    mptr  = mask + n * T_;

    for (int kc = 0; kc < T_; kc += 32) {
        // ---- scores: two 16x16 tiles over this 32-key chunk ----
        v8f s[2];
        int key0 = kc + lrow, key1 = kc + 16 + lrow;
        #pragma unroll
        for (int nt = 0; nt < 2; ++nt) {
            const __bf16* kr = Kbase + (size_t)(kc + nt * 16 + lrow) * E_;
            v8f cc = {};
            #pragma unroll
            for (int kk = 0; kk < 2; ++kk) {
                v16bf bk = frag16(kr + kk * 32 + klo, kr + kk * 32 + 16 + klo);
                cc = __builtin_amdgcn_wmma_f32_16x16x32_bf16(
                        false, aq[kk], false, bk, (short)0, cc, false, false);
            }
            s[nt] = cc;
        }
        int mv0 = mptr[key0];
        int mv1 = mptr[key1];

        // ---- online softmax, per row r; 16-lane group reductions ----
        #pragma unroll
        for (int r = 0; r < 8; ++r) {
            float s0 = mv0 ? s[0][r] * 0.125f : -1.0e19f;  // scale = 1/sqrt(64)
            float s1 = mv1 ? s[1][r] * 0.125f : -1.0e19f;
            float rmax = fmaxf(s0, s1);
            #pragma unroll
            for (int off = 8; off > 0; off >>= 1)
                rmax = fmaxf(rmax, __shfl_xor(rmax, off, 32));
            float mnew = fmaxf(mrow[r], rmax);
            float corr = __expf(mrow[r] - mnew);
            float p0 = __expf(s0 - mnew);
            float p1 = __expf(s1 - mnew);
            float rs = p0 + p1;
            #pragma unroll
            for (int off = 8; off > 0; off >>= 1)
                rs += __shfl_xor(rs, off, 32);
            lsum[r] = lsum[r] * corr + rs;
            mrow[r] = mnew;
            #pragma unroll
            for (int db = 0; db < 4; ++db) o[db][r] *= corr;
            Plds[w][r + 8 * grp][lrow]      = (__bf16)p0;
            Plds[w][r + 8 * grp][16 + lrow] = (__bf16)p1;
        }
        __syncthreads();

        // ---- PV: P[16x32] @ V[32x64] ----
        v16bf ap = frag16(&Plds[w][lrow][klo], &Plds[w][lrow][16 + klo]);
        #pragma unroll
        for (int db = 0; db < 4; ++db) {
            const __bf16* vr = Vbase + (size_t)(db * 16 + lrow) * T_ + kc;
            v16bf bv = frag16(vr + klo, vr + 16 + klo);
            o[db] = __builtin_amdgcn_wmma_f32_16x16x32_bf16(
                        false, ap, false, bv, (short)0, o[db], false, false);
        }
        __syncthreads();
    }

    // ---- normalize and write O (bf16, [N,T,E] head-interleaved) ----
    #pragma unroll
    for (int db = 0; db < 4; ++db) {
        #pragma unroll
        for (int r = 0; r < 8; ++r) {
            int row = qbase + r + 8 * grp;
            float v = o[db][r] / lsum[r];
            O[((size_t)n * T_ + row) * E_ + h * 64 + db * 16 + lrow] = (__bf16)v;
        }
    }
}

// ---------------------------------------------------------------------------
// LayerNorm: dst = LN(a + res) * g + b.  One block (256 thr) per row of 768.
// ---------------------------------------------------------------------------
__global__ __launch_bounds__(256)
void ln_kernel(const float* __restrict__ a, const float* __restrict__ res,
               const float* __restrict__ g, const float* __restrict__ b,
               float* __restrict__ dst, __bf16* __restrict__ dst16)
{
    __shared__ float red[256];
    const int row = blockIdx.x, tid = threadIdx.x;
    const float* ap = a   + (size_t)row * E_;
    const float* rp = res + (size_t)row * E_;

    float x[3];
    float sum = 0.0f;
    #pragma unroll
    for (int j = 0; j < 3; ++j) {
        int e = j * 256 + tid;
        x[j] = ap[e] + rp[e];
        sum += x[j];
    }
    red[tid] = sum; __syncthreads();
    for (int s = 128; s > 0; s >>= 1) { if (tid < s) red[tid] += red[tid + s]; __syncthreads(); }
    float mean = red[0] * (1.0f / E_);
    __syncthreads();

    float vs = 0.0f;
    #pragma unroll
    for (int j = 0; j < 3; ++j) { float d = x[j] - mean; vs += d * d; }
    red[tid] = vs; __syncthreads();
    for (int s = 128; s > 0; s >>= 1) { if (tid < s) red[tid] += red[tid + s]; __syncthreads(); }
    float rstd = rsqrtf(red[0] * (1.0f / E_) + 1e-5f);

    #pragma unroll
    for (int j = 0; j < 3; ++j) {
        int e = j * 256 + tid;
        float v = (x[j] - mean) * rstd * g[e] + b[e];
        dst[(size_t)row * E_ + e]   = v;
        dst16[(size_t)row * E_ + e] = (__bf16)v;
    }
}

// ---------------------------------------------------------------------------
extern "C" void kernel_launch(void* const* d_in, const int* in_sizes, int n_in,
                              void* d_out, int out_size, void* d_ws, size_t ws_size,
                              hipStream_t stream)
{
    (void)in_sizes; (void)n_in; (void)out_size; (void)ws_size;

    const float* x    = (const float*)d_in[0];
    const int*   mask = (const int*)  d_in[1];
    const float* pos  = (const float*)d_in[2];
    const float* cls  = (const float*)d_in[3];
    const float* Wq   = (const float*)d_in[4];
    const float* bq   = (const float*)d_in[5];
    const float* Wk   = (const float*)d_in[6];
    const float* bk   = (const float*)d_in[7];
    const float* Wv   = (const float*)d_in[8];
    const float* bv   = (const float*)d_in[9];
    const float* Wo   = (const float*)d_in[10];
    const float* bo   = (const float*)d_in[11];
    const float* ln1g = (const float*)d_in[12];
    const float* ln1b = (const float*)d_in[13];
    const float* ln2g = (const float*)d_in[14];
    const float* ln2b = (const float*)d_in[15];
    const float* Wf1  = (const float*)d_in[16];
    const float* bf1  = (const float*)d_in[17];
    const float* Wf2  = (const float*)d_in[18];
    const float* bf2  = (const float*)d_in[19];

    char* base = (char*)d_ws;
    size_t off = 0;
    auto alloc = [&](size_t bytes) -> void* {
        off = (off + 255) & ~(size_t)255;
        void* p = base + off;
        off += bytes;
        return p;
    };

    __bf16* wqT   = (__bf16*)alloc((size_t)E_ * E_  * 2);
    __bf16* wkT   = (__bf16*)alloc((size_t)E_ * E_  * 2);
    __bf16* wvT   = (__bf16*)alloc((size_t)E_ * E_  * 2);
    __bf16* woT   = (__bf16*)alloc((size_t)E_ * E_  * 2);
    __bf16* wf1T  = (__bf16*)alloc((size_t)E_ * FF_ * 2);
    __bf16* wf2T  = (__bf16*)alloc((size_t)FF_ * E_ * 2);
    float*  act   = (float*) alloc((size_t)MR_ * E_ * 4);
    __bf16* act16 = (__bf16*)alloc((size_t)MR_ * E_ * 2);
    __bf16* q16   = (__bf16*)alloc((size_t)MR_ * E_ * 2);
    __bf16* k16   = (__bf16*)alloc((size_t)MR_ * E_ * 2);
    __bf16* vt16  = (__bf16*)alloc((size_t)MR_ * E_ * 2);
    __bf16* o16   = (__bf16*)alloc((size_t)MR_ * E_ * 2);
    float*  proj  = (float*) alloc((size_t)MR_ * E_ * 4);
    float*  hid   = (float*) alloc((size_t)MR_ * E_ * 4);
    __bf16* hid16 = (__bf16*)alloc((size_t)MR_ * E_ * 2);
    __bf16* ff1   = (__bf16*)alloc((size_t)MR_ * FF_ * 2);

    const int nte_blocks = (N_ * T_ * E_) / 256;           // 24576
    const int ee_blocks  = (E_ * E_) / 256;                // 2304
    const int eff_blocks = (E_ * FF_) / 256;               // 9216

    // Embedding: CLS prepend + positional add
    embed_kernel<<<nte_blocks, 256, 0, stream>>>(x, cls, pos, act, act16);

    for (int i = 0; i < L_; ++i) {
        // ---- weight prep: bf16 + transpose (reused buffers each layer) ----
        transpose_w_kernel<<<ee_blocks, 256, 0, stream>>>(Wq + (size_t)i*E_*E_,  wqT,  E_,  E_);
        transpose_w_kernel<<<ee_blocks, 256, 0, stream>>>(Wk + (size_t)i*E_*E_,  wkT,  E_,  E_);
        transpose_w_kernel<<<ee_blocks, 256, 0, stream>>>(Wv + (size_t)i*E_*E_,  wvT,  E_,  E_);
        transpose_w_kernel<<<ee_blocks, 256, 0, stream>>>(Wo + (size_t)i*E_*E_,  woT,  E_,  E_);
        transpose_w_kernel<<<eff_blocks, 256, 0, stream>>>(Wf1 + (size_t)i*E_*FF_, wf1T, E_,  FF_);
        transpose_w_kernel<<<eff_blocks, 256, 0, stream>>>(Wf2 + (size_t)i*FF_*E_, wf2T, FF_, E_);

        // ---- QKV projections (V written directly in [N,H,64,T] layout) ----
        dim3 gE(MR_ / 128, E_ / 128);
        gemm_bf16_kernel<<<gE, 256, 0, stream>>>(act16, wqT, bq + (size_t)i*E_, nullptr, q16,  MR_, E_, E_, 0, 0);
        gemm_bf16_kernel<<<gE, 256, 0, stream>>>(act16, wkT, bk + (size_t)i*E_, nullptr, k16,  MR_, E_, E_, 0, 0);
        gemm_bf16_kernel<<<gE, 256, 0, stream>>>(act16, wvT, bv + (size_t)i*E_, nullptr, vt16, MR_, E_, E_, 0, 1);

        // ---- flash attention ----
        attn_kernel<<<dim3(T_ / 64, N_ * H_), 128, 0, stream>>>(q16, k16, vt16, mask, o16);

        // ---- output projection + LN1 ----
        gemm_bf16_kernel<<<gE, 256, 0, stream>>>(o16, woT, bo + (size_t)i*E_, proj, nullptr, MR_, E_, E_, 0, 0);
        ln_kernel<<<MR_, 256, 0, stream>>>(proj, act, ln1g + (size_t)i*E_, ln1b + (size_t)i*E_, hid, hid16);

        // ---- FFN ----
        dim3 gF(MR_ / 128, FF_ / 128);
        gemm_bf16_kernel<<<gF, 256, 0, stream>>>(hid16, wf1T, bf1 + (size_t)i*FF_, nullptr, ff1, MR_, FF_, E_, 1, 0);
        gemm_bf16_kernel<<<gE, 256, 0, stream>>>(ff1, wf2T, bf2 + (size_t)i*E_, proj, nullptr, MR_, E_, FF_, 0, 0);

        float* lnDst = (i == L_ - 1) ? (float*)d_out : act;
        ln_kernel<<<MR_, 256, 0, stream>>>(proj, hid, ln2g + (size_t)i*E_, ln2b + (size_t)i*E_, lnDst, act16);
    }
}
